// arw_FoldingNet_44985487458900
// MI455X (gfx1250) — matrix operand order repaired and checked by reading, not compiled, not run on hardware
//
#include <hip/hip_runtime.h>
#include <hip/hip_bf16.h>

// FoldingNet forward for MI455X (gfx1250, wave32, WMMA).
// B=16, N=8192, K=16, H3=64, H1=512. All fp32 I/O; GEMMs run on
// v_wmma_f32_16x16x32_bf16 with fp32 accumulation. Weights are pre-permuted
// into WMMA B-fragment order in global memory; A is staged fragment-major in
// LDS (b64 stores / b128 loads) to keep DS instruction count minimal.

#define BB  16
#define NN  8192
#define MM  (BB * NN)          // 131072 nodes
#define KNN 16

typedef __attribute__((ext_vector_type(16))) __bf16 v16bf;
typedef __attribute__((ext_vector_type(4)))  __bf16 v4bf;
typedef __attribute__((ext_vector_type(8)))  float  v8f;

// ---------------------------------------------------------------- utilities

__global__ void zero_kernel(float* __restrict__ p, int n) {
  int i = blockIdx.x * 256 + threadIdx.x;
  if (i < n) p[i] = 0.f;
}

// fp32 weights [cin, cout] -> bf16 in WMMA B-fragment order:
// o[((kt*(cout/16) + ct)*32 + lane)*16 + e] = W[kt*32 + (lane>>4)*16 + e][ct*16 + (lane&15)]
// (rows >= cin zero-padded up to cinp)
__global__ void prep_wfrag_kernel(const float* __restrict__ w, __bf16* __restrict__ o,
                                  int cin, int cinp, int cout) {
  int idx = blockIdx.x * 256 + threadIdx.x;
  if (idx >= cinp * cout) return;
  int ctn  = cout >> 4;
  int e    = idx & 15;
  int lane = (idx >> 4) & 31;
  int ct   = (idx >> 9) % ctn;
  int kt   = idx / (512 * ctn);
  int k    = kt * 32 + ((lane >> 4) << 4) + e;
  int c    = (ct << 4) + (lane & 15);
  o[idx] = (__bf16)((k < cin) ? w[k * cout + c] : 0.f);
}

// per-channel sum / sumsq over [M, C]; grid=(ceil(C/64), slices), block=(64,4)
__global__ void stats_kernel(const float* __restrict__ t, float* __restrict__ stats,
                             long M, int C) {
  int c = blockIdx.x * 64 + threadIdx.x;
  float s = 0.f, q = 0.f;
  if (c < C) {
    for (long m = (long)blockIdx.y * 4 + threadIdx.y; m < M; m += (long)gridDim.y * 4) {
      float v = t[m * (long)C + c];
      s += v; q += v * v;
    }
  }
  __shared__ float ss[4][64];
  __shared__ float qq[4][64];
  ss[threadIdx.y][threadIdx.x] = s;
  qq[threadIdx.y][threadIdx.x] = q;
  __syncthreads();
  if (threadIdx.y == 0 && c < C) {
    s = ss[0][threadIdx.x] + ss[1][threadIdx.x] + ss[2][threadIdx.x] + ss[3][threadIdx.x];
    q = qq[0][threadIdx.x] + qq[1][threadIdx.x] + qq[2][threadIdx.x] + qq[3][threadIdx.x];
    atomicAdd(&stats[c], s);
    atomicAdd(&stats[C + c], q);
  }
}

// fold BN(training) into per-channel affine: AB[c]=scale, AB[C+c]=shift
__global__ void affine_kernel(const float* __restrict__ stats,
                              const float* __restrict__ g, const float* __restrict__ be,
                              float* __restrict__ AB, int C, float invCnt) {
  int c = blockIdx.x * blockDim.x + threadIdx.x;
  if (c >= C) return;
  float mean = stats[c] * invCnt;
  float var  = stats[C + c] * invCnt - mean * mean;   // biased variance
  float a = g[c] * rsqrtf(var + 1e-5f);
  AB[c]     = a;
  AB[C + c] = be[c] - mean * a;
}

// -------------------------------------------------- stage 1: cov features

__global__ void cov_kernel(const float* __restrict__ data, const int* __restrict__ knn,
                           float* __restrict__ x12) {
  long m = (long)blockIdx.x * 256 + threadIdx.x;
  if (m >= MM) return;
  int b = (int)(m >> 13);
  const float* db = data + (((long)b) << 13) * 3;
  const int* kn = knn + m * KNN;
  float px[KNN], py[KNN], pz[KNN];
  float sx = 0.f, sy = 0.f, sz = 0.f;
#pragma unroll
  for (int j = 0; j < KNN; ++j) {
    const float* p = db + (long)kn[j] * 3;
    px[j] = p[0]; py[j] = p[1]; pz[j] = p[2];
    sx += px[j]; sy += py[j]; sz += pz[j];
  }
  float mx = sx * (1.f / 16.f), my = sy * (1.f / 16.f), mz = sz * (1.f / 16.f);
  float c00 = 0, c01 = 0, c02 = 0, c11 = 0, c12 = 0, c22 = 0;
#pragma unroll
  for (int j = 0; j < KNN; ++j) {
    float dx = px[j] - mx, dy = py[j] - my, dz = pz[j] - mz;
    c00 += dx * dx; c01 += dx * dy; c02 += dx * dz;
    c11 += dy * dy; c12 += dy * dz; c22 += dz * dz;
  }
  const float s = 1.f / 15.f;   // K-1
  float* o = x12 + m * 12;
  o[0] = data[m * 3 + 0]; o[1] = data[m * 3 + 1]; o[2] = data[m * 3 + 2];
  o[3] = c00 * s; o[4]  = c01 * s; o[5]  = c02 * s;
  o[6] = c01 * s; o[7]  = c11 * s; o[8]  = c12 * s;
  o[9] = c02 * s; o[10] = c12 * s; o[11] = c22 * s;
}

// ----------------------------------------------------- WMMA GEMM (bf16->f32)
// out[M,COUT] = in[M,CIN] @ W + bias, W pre-permuted into B-fragment order.
// Block = 256 thr = 8 waves laid out 2(row-tiles) x 4; each wave owns CT
// column tiles (CT wmma per K-step reusing one A fragment).
// A staged fragment-major in LDS: stage via v4bf (ds_store_b64),
// fragment fetch via v16bf (2x ds_load_b128). B fragments loaded straight
// from global (2x global_load_b128 per tile, 32B/lane coalesced).

template <int CINP, int CIN, int COUT, int CT>
__global__ void __launch_bounds__(256)
gemm_wmma(const float* __restrict__ in, const __bf16* __restrict__ wfrag,
          const float* __restrict__ bias, float* __restrict__ out) {
  __shared__ __align__(32) __bf16 As[2][32][16];   // [row-tile][lane][elem]
  constexpr int CTN = COUT >> 4;                   // total 16-col tiles
  int tid  = threadIdx.x;
  long row0 = (long)blockIdx.x * 32;
  int col0t = blockIdx.y * (4 * CT);               // block's first column tile
  int wave = tid >> 5, lane = tid & 31;
  int tm = wave >> 2, wc = wave & 3;
  int grp = lane >> 4, lr = lane & 15;

  v8f acc[CT];
#pragma unroll
  for (int t = 0; t < CT; ++t) acc[t] = v8f{0.f, 0.f, 0.f, 0.f, 0.f, 0.f, 0.f, 0.f};

  // staging role of this thread: one (row, k-quad) per K-step
  int sr  = tid >> 3;                  // row 0..31
  int sq  = (tid & 7) * 4;             // k-quad base within 32
  int sgrp = sq >> 4;
  int se0  = sq & 15;
  __bf16* sdst = &As[sr >> 4][(sr & 15) + 16 * sgrp][se0];
  const float* srcrow = in + (row0 + sr) * (size_t)CIN;

#pragma unroll
  for (int kt = 0; kt < CINP / 32; ++kt) {
    int kg = kt * 32 + sq;
    if ((kt + 1) * 32 < CIN)
      __builtin_prefetch(srcrow + kg + 32, 0, 1);
    float4 v = make_float4(0.f, 0.f, 0.f, 0.f);
    if (kg + 3 < CIN) {
      v = *(const float4*)(srcrow + kg);
    } else {
      if (kg + 0 < CIN) v.x = srcrow[kg + 0];
      if (kg + 1 < CIN) v.y = srcrow[kg + 1];
      if (kg + 2 < CIN) v.z = srcrow[kg + 2];
    }
    v4bf q4 = {(__bf16)v.x, (__bf16)v.y, (__bf16)v.z, (__bf16)v.w};
    *(v4bf*)sdst = q4;                 // single 8-byte DS store
    __syncthreads();

    v16bf a = *(const v16bf*)(&As[tm][lane][0]);   // 2x ds_load_b128
    const __bf16* wb =
        wfrag + ((size_t)(kt * CTN + col0t + wc * CT) * 32 + lane) * 16;
#pragma unroll
    for (int t = 0; t < CT; ++t) {
      v16bf b = *(const v16bf*)(wb + (size_t)t * 32 * 16);  // 2x global b128
      acc[t] = __builtin_amdgcn_wmma_f32_16x16x32_bf16(false, a, false, b,
                                                       (short)0, acc[t],
                                                       false, false);
    }
    __syncthreads();
  }
  // D layout: VGPR r -> row (grp*8 + r), col = lane%16 within tile
#pragma unroll
  for (int t = 0; t < CT; ++t) {
    int cc = ((col0t + wc * CT + t) << 4) + lr;
    float bsv = bias[cc];
#pragma unroll
    for (int r = 0; r < 8; ++r) {
      long rr = row0 + tm * 16 + grp * 8 + r;
      out[rr * (size_t)COUT + cc] = acc[t][r] + bsv;
    }
  }
}

// ------------------------------------------- GCN aggregation (in input space)
// out[m] = ( relu(aff(t[m])) + sum_j relu(aff(t[knn_j])) ) / 17

template <int C>
__global__ void __launch_bounds__(256)
agg_kernel(const float* __restrict__ t, const float* __restrict__ AB,
           const int* __restrict__ knn, float* __restrict__ out) {
  constexpr int CPT = (C > 256) ? (C / 256) : 1;   // channels per thread
  constexpr int NPB = (C > 256) ? 1 : (256 / C);   // nodes per block
  int tid  = threadIdx.x;
  int slot = (C > 256) ? 0 : (tid / C);
  int cb   = (C > 256) ? tid : (tid % C);
  long m = (long)blockIdx.x * NPB + slot;
  long base = (m >> 13) << 13;                     // batch row base
  const int* kn = knn + m * KNN;
  float A[CPT], Sh[CPT], acc[CPT];
#pragma unroll
  for (int q = 0; q < CPT; ++q) { A[q] = AB[cb + q * 256]; Sh[q] = AB[C + cb + q * 256]; }
  const float* tr = t + m * (long)C;
#pragma unroll
  for (int q = 0; q < CPT; ++q)
    acc[q] = fmaxf(tr[cb + q * 256] * A[q] + Sh[q], 0.f);
  for (int j = 0; j < KNN; ++j) {
    const float* nr = t + (base + kn[j]) * (long)C;
#pragma unroll
    for (int q = 0; q < CPT; ++q)
      acc[q] += fmaxf(nr[cb + q * 256] * A[q] + Sh[q], 0.f);
  }
#pragma unroll
  for (int q = 0; q < CPT; ++q)
    out[m * (long)C + cb + q * 256] = acc[q] * (1.f / 17.f);
}

// -------------------------------------------------------- pool / e2 / decode

__global__ void maxpool_kernel(const float* __restrict__ h2, const float* __restrict__ AB3,
                               float* __restrict__ pooled) {
  int b = blockIdx.x;
  int c = blockIdx.y * 64 + threadIdx.x;
  const float* base = h2 + ((long)b << 13) * 512;
  float a = AB3[c], sh = AB3[512 + c];
  float mx = 0.f;                                  // relu >= 0
  for (int n = threadIdx.y; n < NN; n += 4)
    mx = fmaxf(mx, base[(long)n * 512 + c] * a + sh);
  __shared__ float red[4][64];
  red[threadIdx.y][threadIdx.x] = mx;
  __syncthreads();
  if (threadIdx.y == 0) {
    mx = fmaxf(fmaxf(red[0][threadIdx.x], red[1][threadIdx.x]),
               fmaxf(red[2][threadIdx.x], red[3][threadIdx.x]));
    pooled[b * 512 + c] = mx;                      // max(relu(x)) == relu(max(x))
  }
}

__global__ void e2gemm_kernel(const float* __restrict__ pooled, const float* __restrict__ w,
                              const float* __restrict__ bias, float* __restrict__ t4) {
  int b = blockIdx.x;
  __shared__ float p[512];
  for (int i = threadIdx.x; i < 512; i += 256) p[i] = pooled[b * 512 + i];
  __syncthreads();
  for (int c = threadIdx.x; c < 512; c += 256) {
    float s = bias[c];
    for (int k = 0; k < 512; ++k) s += p[k] * w[k * 512 + c];
    t4[b * 512 + c] = s;
  }
}

__global__ void e2bn_kernel(const float* __restrict__ t4, const float* __restrict__ g,
                            const float* __restrict__ be, float* __restrict__ cw) {
  int c = blockIdx.x * 256 + threadIdx.x;
  if (c >= 512) return;
  float s = 0.f, q = 0.f;
  for (int b = 0; b < BB; ++b) { float v = t4[b * 512 + c]; s += v; q += v * v; }
  float mean = s * (1.f / 16.f);
  float var  = q * (1.f / 16.f) - mean * mean;
  float a = g[c] * rsqrtf(var + 1e-5f);
  float sh = be[c] - mean * a;
  for (int b = 0; b < BB; ++b)
    cw[b * 512 + c] = fmaxf(t4[b * 512 + c] * a + sh, 0.f);
}

// base1[b,c] = cw[b,:].d1_w[:512,c] + d1_b[c];  base2 same with d2_w/d2_b
__global__ void base_kernel(const float* __restrict__ cw,
                            const float* __restrict__ d1w, const float* __restrict__ d1b,
                            const float* __restrict__ d2w, const float* __restrict__ d2b,
                            float* __restrict__ base1, float* __restrict__ base2) {
  int tid = threadIdx.x;
  if (tid >= 96) return;
  int b = tid / 6, q = tid % 6, c = q % 3;
  const float* w = (q < 3) ? d1w : d2w;
  float s = (q < 3) ? d1b[c] : d2b[c];
  for (int k = 0; k < 512; ++k) s += cw[b * 512 + k] * w[k * 3 + c];
  ((q < 3) ? base1 : base2)[b * 3 + c] = s;
}

__global__ void dec1_kernel(const float* __restrict__ base1, const float* __restrict__ d1w,
                            float* __restrict__ z1pre) {
  long m = (long)blockIdx.x * 256 + threadIdx.x;
  int b = (int)(m >> 13);
  int n = (int)(m & 8191);
  int i = n / 65, j = n % 65;                       // nx=129, ny=65 grid
  float gx = 1.f + i * (119.f / 128.f);
  float gy = 1.f + j * (59.f / 64.f);
#pragma unroll
  for (int c = 0; c < 3; ++c)
    z1pre[m * 3 + c] = base1[b * 3 + c] + gx * d1w[512 * 3 + c] + gy * d1w[513 * 3 + c];
}

__global__ void dec2_kernel(const float* __restrict__ z1pre, const float* __restrict__ AB5,
                            const float* __restrict__ base2, const float* __restrict__ d2w,
                            float* __restrict__ z2pre) {
  long m = (long)blockIdx.x * 256 + threadIdx.x;
  int b = (int)(m >> 13);
  float zr[3];
#pragma unroll
  for (int k = 0; k < 3; ++k)
    zr[k] = fmaxf(z1pre[m * 3 + k] * AB5[k] + AB5[3 + k], 0.f);
#pragma unroll
  for (int c = 0; c < 3; ++c) {
    float s = base2[b * 3 + c];
#pragma unroll
    for (int k = 0; k < 3; ++k) s += zr[k] * d2w[(512 + k) * 3 + c];
    z2pre[m * 3 + c] = s;
  }
}

__global__ void final_kernel(const float* __restrict__ z, const float* __restrict__ AB,
                             float* __restrict__ out) {
  int i = blockIdx.x * 256 + threadIdx.x;
  int c = i % 3;
  out[i] = fmaxf(z[i] * AB[c] + AB[3 + c], 0.f);
}

// --------------------------------------------------------------- launcher

extern "C" void kernel_launch(void* const* d_in, const int* in_sizes, int n_in,
                              void* d_out, int out_size, void* d_ws, size_t ws_size,
                              hipStream_t stream) {
  const float* data  = (const float*)d_in[0];
  const int*   knn   = (const int*)  d_in[1];
  const float* e1_w  = (const float*)d_in[2];
  const float* e1_b  = (const float*)d_in[3];
  const float* e1_g  = (const float*)d_in[4];
  const float* e1_be = (const float*)d_in[5];
  const float* gc1_w = (const float*)d_in[6];
  const float* gc1_b = (const float*)d_in[7];
  const float* g1_g  = (const float*)d_in[8];
  const float* g1_be = (const float*)d_in[9];
  const float* gc2_w = (const float*)d_in[10];
  const float* gc2_b = (const float*)d_in[11];
  const float* g2_g  = (const float*)d_in[12];
  const float* g2_be = (const float*)d_in[13];
  const float* e2_w  = (const float*)d_in[14];
  const float* e2_b  = (const float*)d_in[15];
  const float* e2_g  = (const float*)d_in[16];
  const float* e2_be = (const float*)d_in[17];
  const float* d1_w  = (const float*)d_in[18];
  const float* d1_b  = (const float*)d_in[19];
  const float* d1_g  = (const float*)d_in[20];
  const float* d1_be = (const float*)d_in[21];
  const float* d2_w  = (const float*)d_in[22];
  const float* d2_b  = (const float*)d_in[23];
  const float* d2_g  = (const float*)d_in[24];
  const float* d2_be = (const float*)d_in[25];

  char* ws = (char*)d_ws;
  size_t off = 0;
  auto take = [&](size_t bytes) -> char* {
    char* p = ws + off;
    off += (bytes + 255) & ~(size_t)255;
    return p;
  };
  float*  h2     = (float*) take((size_t)MM * 512 * 4);
  float*  h1     = (float*) take((size_t)MM * 128 * 4);
  float*  x128a  = (float*) take((size_t)MM * 128 * 4);
  float*  t1     = (float*) take((size_t)MM * 64 * 4);
  float*  x64a   = (float*) take((size_t)MM * 64 * 4);
  float*  x12    = (float*) take((size_t)MM * 12 * 4);
  float*  z1pre  = (float*) take((size_t)MM * 3 * 4);
  float*  z2pre  = (float*) take((size_t)MM * 3 * 4);
  __bf16* w1bf   = (__bf16*)take(32 * 64 * 2);
  __bf16* w2bf   = (__bf16*)take(64 * 128 * 2);
  __bf16* w3bf   = (__bf16*)take(128 * 512 * 2);
  float*  pooled = (float*) take(16 * 512 * 4);
  float*  t4     = (float*) take(16 * 512 * 4);
  float*  cw     = (float*) take(16 * 512 * 4);
  float*  base1  = (float*) take(16 * 3 * 4);
  float*  base2  = (float*) take(16 * 3 * 4);
  float*  stats  = (float*) take(2048 * 4);
  float *st1 = stats, *st2 = stats + 128, *st3 = stats + 384;
  float *st5 = stats + 1408, *st6 = stats + 1440;
  float* AB1 = (float*)take(128 * 4);
  float* AB2 = (float*)take(256 * 4);
  float* AB3 = (float*)take(1024 * 4);
  float* AB5 = (float*)take(8 * 4);
  float* AB6 = (float*)take(8 * 4);

  const float invM = 1.f / (float)MM;

  zero_kernel<<<8, 256, 0, stream>>>(stats, 2048);
  prep_wfrag_kernel<<<(32 * 64 + 255) / 256, 256, 0, stream>>>(e1_w, w1bf, 12, 32, 64);
  prep_wfrag_kernel<<<(64 * 128 + 255) / 256, 256, 0, stream>>>(gc1_w, w2bf, 64, 64, 128);
  prep_wfrag_kernel<<<(128 * 512 + 255) / 256, 256, 0, stream>>>(gc2_w, w3bf, 128, 128, 512);

  cov_kernel<<<MM / 256, 256, 0, stream>>>(data, knn, x12);

  gemm_wmma<32, 12, 64, 1><<<dim3(MM / 32, 1), 256, 0, stream>>>(x12, w1bf, e1_b, t1);
  stats_kernel<<<dim3(1, 256), dim3(64, 4), 0, stream>>>(t1, st1, MM, 64);
  affine_kernel<<<1, 64, 0, stream>>>(st1, e1_g, e1_be, AB1, 64, invM);
  agg_kernel<64><<<MM / 4, 256, 0, stream>>>(t1, AB1, knn, x64a);

  gemm_wmma<64, 64, 128, 2><<<dim3(MM / 32, 1), 256, 0, stream>>>(x64a, w2bf, gc1_b, h1);
  stats_kernel<<<dim3(2, 256), dim3(64, 4), 0, stream>>>(h1, st2, MM, 128);
  affine_kernel<<<1, 128, 0, stream>>>(st2, g1_g, g1_be, AB2, 128, invM);
  agg_kernel<128><<<MM / 2, 256, 0, stream>>>(h1, AB2, knn, x128a);

  gemm_wmma<128, 128, 512, 2><<<dim3(MM / 32, 4), 256, 0, stream>>>(x128a, w3bf, gc2_b, h2);
  stats_kernel<<<dim3(8, 256), dim3(64, 4), 0, stream>>>(h2, st3, MM, 512);
  affine_kernel<<<2, 256, 0, stream>>>(st3, g2_g, g2_be, AB3, 512, invM);

  maxpool_kernel<<<dim3(16, 8), dim3(64, 4), 0, stream>>>(h2, AB3, pooled);
  e2gemm_kernel<<<16, 256, 0, stream>>>(pooled, e2_w, e2_b, t4);
  e2bn_kernel<<<2, 256, 0, stream>>>(t4, e2_g, e2_be, cw);

  base_kernel<<<1, 128, 0, stream>>>(cw, d1_w, d1_b, d2_w, d2_b, base1, base2);
  dec1_kernel<<<MM / 256, 256, 0, stream>>>(base1, d1_w, z1pre);
  stats_kernel<<<dim3(1, 256), dim3(64, 4), 0, stream>>>(z1pre, st5, MM, 3);
  affine_kernel<<<1, 64, 0, stream>>>(st5, d1_g, d1_be, AB5, 3, invM);
  dec2_kernel<<<MM / 256, 256, 0, stream>>>(z1pre, AB5, base2, d2_w, z2pre);
  stats_kernel<<<dim3(1, 256), dim3(64, 4), 0, stream>>>(z2pre, st6, MM, 3);
  affine_kernel<<<1, 64, 0, stream>>>(st6, d2_g, d2_be, AB6, 3, invM);
  final_kernel<<<(MM * 3) / 256, 256, 0, stream>>>(z2pre, AB6, (float*)d_out);

  (void)in_sizes; (void)n_in; (void)out_size; (void)ws_size;
}
